// MultiHeadAttention_30631706755353
// MI455X (gfx1250) — matrix operand order repaired and checked by reading, not compiled, not run on hardware
//
#include <hip/hip_runtime.h>

// ---------------------------------------------------------------------------
// Causal MHA forward for MI455X (gfx1250, wave32, WMMA bf16 -> fp32 accum).
//   k_cvt  x5 : fp32 -> bf16 (x, Wq, Wk, Wv, Wo)
//   k_qkv     : Q/K/V = x @ W^T; weight panel async-staged to LDS (double
//               buffered, GLOBAL_LOAD_ASYNC_TO_LDS_B128 + s_wait_asynccnt),
//               shared by all 8 waves; 16x64 output per wave
//   k_vt      : tiled transpose of V (bf16)
//   k_attn    : FUSED scores + softmax + P@V per (head, 16-query strip);
//               131KB fp32 score strip in LDS, attn written to HBM once
//   k_oproj   : out = ctx @ Wo^T, same LDS-staged GEMM, fp32 out
// ---------------------------------------------------------------------------

#define MODEL_DIM 1024
#define NUM_HEADS 16
#define HEAD_DIM  64
#define SEQ       2048
#define SSTRIDE   2056                       // 2048 + 8 pad (16B-aligned rows)
#define SMEM_FLOATS (16 * SSTRIDE + 8 * 256)
#define SMEM_BYTES  (SMEM_FLOATS * 4)        // 139,776 B
#define PSTRIDE   40                         // LDS weight-panel row stride (bf16)

typedef __attribute__((ext_vector_type(16))) __bf16 v16bf;
typedef __attribute__((ext_vector_type(2)))  __bf16 v2bf;
typedef __attribute__((ext_vector_type(8)))  float  v8f;
typedef int v4i __attribute__((vector_size(16)));   // builtin's expected elt type

union FragBF {
  v16bf v;
  unsigned u[8];   // dword i = frag elements 2i (lo), 2i+1 (hi)
};

__device__ __forceinline__ unsigned pkbf(float lo, float hi) {
  v2bf v = {(__bf16)lo, (__bf16)hi};        // -> v_cvt_pk_bf16_f32
  return __builtin_bit_cast(unsigned, v);
}
__device__ __forceinline__ unsigned short bf1(float f) {
  return __builtin_bit_cast(unsigned short, (__bf16)f);
}

__device__ __forceinline__ v8f wmma_bf16(v16bf a, v16bf b, v8f c) {
  return __builtin_amdgcn_wmma_f32_16x16x32_bf16(false, a, false, b, (short)0,
                                                 c, false, false);
}

// ---- CDNA5 async global->LDS copy (ASYNCcnt), with safe fallback ----------
#if defined(__AMDGCN__) && __has_builtin(__builtin_amdgcn_global_load_async_to_lds_b128)
#define HAS_ASYNC_LDS 1
#endif

__device__ __forceinline__ void async_copy_b128(const unsigned short* g,
                                                unsigned short* l) {
#ifdef HAS_ASYNC_LDS
  __builtin_amdgcn_global_load_async_to_lds_b128(
      (__attribute__((address_space(1))) v4i*)g,
      (__attribute__((address_space(3))) v4i*)l, 0, 0);
#else
  *(uint4*)l = *(const uint4*)g;
#endif
}
__device__ __forceinline__ void wait_async() {
#ifdef HAS_ASYNC_LDS
#if __has_builtin(__builtin_amdgcn_s_wait_asynccnt)
  __builtin_amdgcn_s_wait_asynccnt(0);
#else
  asm volatile("s_wait_asynccnt 0x0" ::: "memory");
#endif
#endif
}

// ---- Fragment builders (contiguous k = 16*(lane/16) + s, wave32) -------------------------
__device__ __forceinline__ v16bf afrag_bf16(const unsigned short* __restrict__ src,
                                            int stride, int m0, int k0, int lane) {
  int row = m0 + (lane & 15);
  int kb  = k0 + ((lane & 16) ? 16 : 0);
  const unsigned short* p = src + (long)row * stride + kb;
  uint4 c0 = *(const uint4*)(p);
  uint4 c1 = *(const uint4*)(p + 8);
  FragBF f;
  f.u[0] = c0.x; f.u[1] = c0.y; f.u[2] = c0.z; f.u[3] = c0.w;
  f.u[4] = c1.x; f.u[5] = c1.y; f.u[6] = c1.z; f.u[7] = c1.w;
  return f.v;
}

__device__ __forceinline__ v16bf afrag_f32(const float* __restrict__ src,
                                           int stride, int m0, int k0, int lane) {
  int row = m0 + (lane & 15);
  int kb  = k0 + ((lane & 16) ? 16 : 0);
  const float* p = src + (long)row * stride + kb;
  float4 c0 = *(const float4*)(p);
  float4 c1 = *(const float4*)(p + 4);
  float4 c2 = *(const float4*)(p + 8);
  float4 c3 = *(const float4*)(p + 12);
  FragBF f;
  f.u[0] = pkbf(c0.x, c0.y); f.u[1] = pkbf(c0.z, c0.w);
  f.u[2] = pkbf(c1.x, c1.y); f.u[3] = pkbf(c1.z, c1.w);
  f.u[4] = pkbf(c2.x, c2.y); f.u[5] = pkbf(c2.z, c2.w);
  f.u[6] = pkbf(c3.x, c3.y); f.u[7] = pkbf(c3.z, c3.w);
  return f.v;
}

// B[k][n] = src[n*stride + k]  (global, contiguous in k)
__device__ __forceinline__ v16bf bfragT_bf16(const unsigned short* __restrict__ src,
                                             int stride, int n0, int k0, int lane) {
  int col = n0 + (lane & 15);
  int kb  = k0 + ((lane & 16) ? 16 : 0);
  const uint4* p = (const uint4*)(src + (long)col * stride + kb);
  uint4 c0 = p[0];
  uint4 c1 = p[1];
  FragBF f;
  f.u[0] = c0.x; f.u[1] = c0.y; f.u[2] = c0.z; f.u[3] = c0.w;
  f.u[4] = c1.x; f.u[5] = c1.y; f.u[6] = c1.z; f.u[7] = c1.w;
  return f.v;
}

// B-frag from LDS weight panel: panel[col][k], row stride PSTRIDE
__device__ __forceinline__ v16bf bfragT_lds(const unsigned short* buf,
                                            int nloc, int lane) {
  int col = nloc + (lane & 15);
  int kb  = (lane & 16) ? 16 : 0;
  const unsigned short* p = buf + col * PSTRIDE + kb;
  uint4 c0 = *(const uint4*)(p);
  uint4 c1 = *(const uint4*)(p + 8);
  FragBF f;
  f.u[0] = c0.x; f.u[1] = c0.y; f.u[2] = c0.z; f.u[3] = c0.w;
  f.u[4] = c1.x; f.u[5] = c1.y; f.u[6] = c1.z; f.u[7] = c1.w;
  return f.v;
}

// stage one 64(n) x 32(k) bf16 weight panel into LDS: 256 threads x 16B
__device__ __forceinline__ void stage_panel(const unsigned short* __restrict__ W,
                                            unsigned short* buf, int n0, int k0,
                                            int tid) {
  int row = tid >> 2, ch = tid & 3;
  async_copy_b128(W + (long)(n0 + row) * MODEL_DIM + k0 + ch * 8,
                  buf + row * PSTRIDE + ch * 8);
}

// ---- fp32 -> bf16 bulk convert --------------------------------------------
__global__ __launch_bounds__(256) void k_cvt(const float* __restrict__ src,
                                             unsigned short* __restrict__ dst,
                                             int n4) {
  int i = blockIdx.x * 256 + threadIdx.x;
  if (i >= n4) return;
  float4 f = ((const float4*)src)[i];
  uint2 o;
  o.x = pkbf(f.x, f.y);
  o.y = pkbf(f.z, f.w);
  ((uint2*)dst)[i] = o;
}

// ---- Q/K/V projection: block = 128m x 64n, weight panel via async LDS -----
__global__ __launch_bounds__(256) void k_qkv(
    const unsigned short* __restrict__ xb,
    const unsigned short* __restrict__ Wqb, const unsigned short* __restrict__ Wkb,
    const unsigned short* __restrict__ Wvb,
    unsigned short* __restrict__ Qb, unsigned short* __restrict__ Kb,
    unsigned short* __restrict__ Vb) {
  __shared__ unsigned short bpanel[2][64 * PSTRIDE];
  const unsigned short* W =
      (blockIdx.z == 0) ? Wqb : (blockIdx.z == 1) ? Wkb : Wvb;
  unsigned short* out = (blockIdx.z == 0) ? Qb : (blockIdx.z == 1) ? Kb : Vb;
  int tid = threadIdx.x, wave = tid >> 5, lane = tid & 31;
  int m0 = blockIdx.x * 128 + wave * 16;
  int n0 = blockIdx.y * 64;

  stage_panel(W, bpanel[0], n0, 0, tid);
  v8f acc[4] = {};
  for (int k0 = 0; k0 < MODEL_DIM; k0 += 32) {
    int cur = (k0 >> 5) & 1;
    wait_async();
    __syncthreads();                        // panel[cur] ready for everyone
    if (k0 + 32 < MODEL_DIM) stage_panel(W, bpanel[cur ^ 1], n0, k0 + 32, tid);
    v16bf a = afrag_bf16(xb, MODEL_DIM, m0, k0, lane);
#pragma unroll
    for (int j = 0; j < 4; ++j)
      acc[j] = wmma_bf16(a, bfragT_lds(bpanel[cur], 16 * j, lane), acc[j]);
  }
  int m = m0 + ((lane & 16) ? 8 : 0), n = lane & 15;
#pragma unroll
  for (int j = 0; j < 4; ++j)
#pragma unroll
    for (int r = 0; r < 8; ++r)
      out[(long)(m + r) * MODEL_DIM + n0 + 16 * j + n] = bf1(acc[j][r]);
}

// ---- tiled transpose: Vt[c][k] = V[k][c] ----------------------------------
__global__ __launch_bounds__(256) void k_vt(const unsigned short* __restrict__ Vb,
                                            unsigned short* __restrict__ Vt) {
  __shared__ unsigned short tile[32][33];
  int tx = threadIdx.x & 31, ty = threadIdx.x >> 5;
  int k0 = blockIdx.x * 32, c0 = blockIdx.y * 32;
#pragma unroll
  for (int i = ty; i < 32; i += 8)
    tile[i][tx] = Vb[(long)(k0 + i) * MODEL_DIM + c0 + tx];
  __syncthreads();
#pragma unroll
  for (int i = ty; i < 32; i += 8)
    Vt[(long)(c0 + i) * SEQ + k0 + tx] = tile[tx][i];
}

// ---- FUSED: scores -> softmax -> P@V for one (head, 16-query strip) -------
__global__ __launch_bounds__(256) void k_attn(
    const unsigned short* __restrict__ Qb, const unsigned short* __restrict__ Kb,
    const unsigned short* __restrict__ Vt, float* __restrict__ attn,
    unsigned short* __restrict__ Cb) {
  extern __shared__ float smem[];
  float* strip = smem;                     // [16][SSTRIDE] scores / probs
  float* part  = smem + 16 * SSTRIDE;      // [8][256] ctx partials

  int wave = threadIdx.x >> 5, lane = threadIdx.x & 31;
  int qT = blockIdx.x, h = blockIdx.y;
  int q0 = qT << 4;
  int hc = h * HEAD_DIM;
  int m  = ((lane & 16) ? 8 : 0), n = lane & 15;
  int cmax = q0 + 16;
  int cpad = (cmax + 31) & ~31;

  // stage A: live score tiles (cT <= qT) into LDS
  v16bf aq0 = afrag_bf16(Qb, MODEL_DIM, q0, hc, lane);
  v16bf aq1 = afrag_bf16(Qb, MODEL_DIM, q0, hc + 32, lane);
  for (int cT = wave; cT <= qT; cT += 8) {
    int c0 = cT << 4;
    v8f acc = {};
    acc = wmma_bf16(aq0, bfragT_bf16(Kb, MODEL_DIM, c0, hc, lane), acc);
    acc = wmma_bf16(aq1, bfragT_bf16(Kb, MODEL_DIM, c0, hc + 32, lane), acc);
#pragma unroll
    for (int r = 0; r < 8; ++r)
      strip[(m + r) * SSTRIDE + c0 + n] = acc[r] * 0.125f;
  }
  __syncthreads();

  // stage B: softmax (16 threads/row, float4 chunks); single HBM attn write
  {
    int row = threadIdx.x >> 4;
    int lc  = threadIdx.x & 15;
    int q   = q0 + row;
    float* srow = strip + row * SSTRIDE;
    float* arow = attn + ((long)h * SEQ + q) * SEQ;
    const float NI = -__builtin_inff();

    float mx = NI;
    for (int c = lc * 4; c < cmax; c += 64) {
      float4 v = *(const float4*)(srow + c);
      mx = fmaxf(mx, (c + 0 <= q) ? v.x : NI);
      mx = fmaxf(mx, (c + 1 <= q) ? v.y : NI);
      mx = fmaxf(mx, (c + 2 <= q) ? v.z : NI);
      mx = fmaxf(mx, (c + 3 <= q) ? v.w : NI);
    }
#pragma unroll
    for (int o = 8; o > 0; o >>= 1) mx = fmaxf(mx, __shfl_xor(mx, o, 16));

    float s = 0.f;
    for (int c = lc * 4; c < cpad; c += 64) {
      float4 e = {0.f, 0.f, 0.f, 0.f};
      if (c < cmax) {
        float4 v = *(const float4*)(srow + c);
        e.x = (c + 0 <= q) ? __expf(v.x - mx) : 0.f;
        e.y = (c + 1 <= q) ? __expf(v.y - mx) : 0.f;
        e.z = (c + 2 <= q) ? __expf(v.z - mx) : 0.f;
        e.w = (c + 3 <= q) ? __expf(v.w - mx) : 0.f;
      }
      *(float4*)(srow + c) = e;
      s += e.x + e.y + e.z + e.w;
    }
#pragma unroll
    for (int o = 8; o > 0; o >>= 1) s += __shfl_xor(s, o, 16);
    float inv = 1.0f / s;

    for (int c = lc * 4; c < cpad; c += 64) {
      float4 v = *(float4*)(srow + c);
      v.x *= inv; v.y *= inv; v.z *= inv; v.w *= inv;
      *(float4*)(srow + c) = v;            // normalized P for stage C
      *(float4*)(arow + c) = v;            // attn output
    }
  }
  // flat float4 zero-fill of the masked tail [cpad, SEQ) across all threads
  {
    int nz4 = (SEQ - cpad) >> 2;
    int total = 16 * nz4;
    float4 z = {0.f, 0.f, 0.f, 0.f};
    for (int i = threadIdx.x; i < total; i += 256) {
      int row = i / nz4, cc = cpad + ((i - row * nz4) << 2);
      *(float4*)(attn + ((long)h * SEQ + q0 + row) * SEQ + cc) = z;
    }
  }
  __syncthreads();

  // stage C: ctx = P @ V, 4 n-tiles x 2 K-halves across 8 waves
  {
    int nT = wave & 3, half = wave >> 2;
    int ns = cpad >> 5;
    v8f acc = {};
    for (int s2 = half; s2 < ns; s2 += 2) {
      int k0 = s2 << 5;
      v16bf a = afrag_f32(strip, SSTRIDE, 0, k0, lane);
      v16bf b = bfragT_bf16(Vt, SEQ, hc + (nT << 4), k0, lane);
      acc = wmma_bf16(a, b, acc);
    }
#pragma unroll
    for (int r = 0; r < 8; ++r) part[wave * 256 + lane * 8 + r] = acc[r];
  }
  __syncthreads();
  if (wave < 4) {
#pragma unroll
    for (int r = 0; r < 8; ++r) {
      float v = part[wave * 256 + lane * 8 + r] +
                part[(wave + 4) * 256 + lane * 8 + r];
      Cb[(long)(q0 + m + r) * MODEL_DIM + hc + (wave << 4) + n] = bf1(v);
    }
  }
}

// ---- output projection: same LDS-staged GEMM, fp32 out --------------------
__global__ __launch_bounds__(256) void k_oproj(
    const unsigned short* __restrict__ Cb, const unsigned short* __restrict__ Wob,
    float* __restrict__ out) {
  __shared__ unsigned short bpanel[2][64 * PSTRIDE];
  int tid = threadIdx.x, wave = tid >> 5, lane = tid & 31;
  int m0 = blockIdx.x * 128 + wave * 16;
  int n0 = blockIdx.y * 64;

  stage_panel(Wob, bpanel[0], n0, 0, tid);
  v8f acc[4] = {};
  for (int k0 = 0; k0 < MODEL_DIM; k0 += 32) {
    int cur = (k0 >> 5) & 1;
    wait_async();
    __syncthreads();
    if (k0 + 32 < MODEL_DIM) stage_panel(Wob, bpanel[cur ^ 1], n0, k0 + 32, tid);
    v16bf a = afrag_bf16(Cb, MODEL_DIM, m0, k0, lane);
#pragma unroll
    for (int j = 0; j < 4; ++j)
      acc[j] = wmma_bf16(a, bfragT_lds(bpanel[cur], 16 * j, lane), acc[j]);
  }
  int m = m0 + ((lane & 16) ? 8 : 0), n = lane & 15;
#pragma unroll
  for (int j = 0; j < 4; ++j)
#pragma unroll
    for (int r = 0; r < 8; ++r)
      out[(long)(m + r) * MODEL_DIM + n0 + 16 * j + n] = acc[j][r];
}

// ---------------------------------------------------------------------------
extern "C" void kernel_launch(void* const* d_in, const int* in_sizes, int n_in,
                              void* d_out, int out_size, void* d_ws, size_t ws_size,
                              hipStream_t stream) {
  const float* x  = (const float*)d_in[0];
  const float* Wq = (const float*)d_in[1];
  const float* Wk = (const float*)d_in[2];
  const float* Wv = (const float*)d_in[3];
  const float* Wo = (const float*)d_in[4];
  // d_in[5]: causal mask — computed analytically.

  float* out  = (float*)d_out;                            // [2048,1024]
  float* attn = (float*)d_out + (long)SEQ * MODEL_DIM;    // [16,2048,2048]

  unsigned short* xb  = (unsigned short*)d_ws;            // 32 MB total
  unsigned short* Wqb = xb  + (long)SEQ * MODEL_DIM;
  unsigned short* Wkb = Wqb + (long)MODEL_DIM * MODEL_DIM;
  unsigned short* Wvb = Wkb + (long)MODEL_DIM * MODEL_DIM;
  unsigned short* Wob = Wvb + (long)MODEL_DIM * MODEL_DIM;
  unsigned short* Qb  = Wob + (long)MODEL_DIM * MODEL_DIM;
  unsigned short* Kb  = Qb  + (long)SEQ * MODEL_DIM;
  unsigned short* Vb  = Kb  + (long)SEQ * MODEL_DIM;
  unsigned short* Vt  = Vb  + (long)SEQ * MODEL_DIM;
  unsigned short* Cb  = Vt  + (long)SEQ * MODEL_DIM;

  const int n4x = SEQ * MODEL_DIM / 4;
  const int n4w = MODEL_DIM * MODEL_DIM / 4;
  k_cvt<<<dim3(n4x / 256), 256, 0, stream>>>(x,  xb,  n4x);
  k_cvt<<<dim3(n4w / 256), 256, 0, stream>>>(Wq, Wqb, n4w);
  k_cvt<<<dim3(n4w / 256), 256, 0, stream>>>(Wk, Wkb, n4w);
  k_cvt<<<dim3(n4w / 256), 256, 0, stream>>>(Wv, Wvb, n4w);
  k_cvt<<<dim3(n4w / 256), 256, 0, stream>>>(Wo, Wob, n4w);

  k_qkv  <<<dim3(16, 16, 3), 256, 0, stream>>>(xb, Wqb, Wkb, Wvb, Qb, Kb, Vb);
  k_vt   <<<dim3(SEQ / 32, MODEL_DIM / 32), 256, 0, stream>>>(Vb, Vt);
  k_attn <<<dim3(SEQ / 16, NUM_HEADS), 256, SMEM_BYTES, stream>>>(Qb, Kb, Vt,
                                                                  attn, Cb);
  k_oproj<<<dim3(16, 16), 256, 0, stream>>>(Cb, Wob, out);
}
